// MultiHeadAttention_17952963297604
// MI455X (gfx1250) — matrix operand order repaired and checked by reading, not compiled, not run on hardware
//
#include <hip/hip_runtime.h>

// ---------------------------------------------------------------------------
// MI455X (gfx1250) multi-head attention forward.
// bf16 WMMA + fp32 accumulate; async global->LDS, double-buffered pipelines.
// ---------------------------------------------------------------------------

typedef __bf16 bf16_t;
typedef bf16_t v16bf __attribute__((ext_vector_type(16)));
typedef float  v8f   __attribute__((ext_vector_type(8)));

#define D_MODEL 512
#define SEQ     4096
#define NHEAD   8
#define DEPTH   64
#define BATCH   2

__device__ __forceinline__ unsigned short f32_to_bf16(float f) {
  unsigned int u = __float_as_uint(f);
  u += 0x7FFFu + ((u >> 16) & 1u);           // round-to-nearest-even
  return (unsigned short)(u >> 16);
}
__device__ __forceinline__ unsigned int pack_bf16x2(float a, float b) {
  return (unsigned int)f32_to_bf16(a) | ((unsigned int)f32_to_bf16(b) << 16);
}

struct FragBF { union { v16bf v; uint4 q[2]; }; };

__device__ __forceinline__ v8f wmma_bf16(const FragBF& a, const FragBF& b, v8f c) {
  // v_wmma_f32_16x16x32_bf16: (neg_a, A, neg_b, B, c_mod, C, reuse_a, reuse_b)
  return __builtin_amdgcn_wmma_f32_16x16x32_bf16(false, a.v, false, b.v,
                                                 (short)0, c, false, false);
}

// A fragment (16x32 bf16, ISA 7.12.2):
//  lane<16: row=lane,     K = {0..7} then {16..23}
//  lane>=16: row=lane-16, K = {8..15} then {24..31}
__device__ __forceinline__ FragBF load_frag_a(const unsigned short* base, int stride, int lane) {
  FragBF f;
  const unsigned short* p = base + (lane & 15) * stride + ((lane >> 4) << 3);
  f.q[0] = *reinterpret_cast<const uint4*>(p);
  f.q[1] = *reinterpret_cast<const uint4*>(p + 16);
  return f;
}
// B fragment (32x16 bf16): lane holds column = lane&15, K = {0..7,16..23} (lane<16) / {8..15,24..31}.
__device__ __forceinline__ FragBF load_frag_b(const unsigned short* base, int stride, int lane) {
  FragBF f;
  unsigned short* e = reinterpret_cast<unsigned short*>(&f);
  const unsigned short* p = base + (lane & 15) + ((lane >> 4) << 3) * stride;
  #pragma unroll
  for (int s = 0; s < 8; ++s) {
    e[s]     = p[s * stride];
    e[s + 8] = p[(s + 16) * stride];
  }
  return f;
}

// ---------------------------------------------------------------------------
// CDNA5 async global->LDS (ASYNCcnt-tracked, bypasses VGPRs).
// Generic pointers to __shared__ carry the LDS byte address in their low
// 32 bits on gfx1250.
// ---------------------------------------------------------------------------
__device__ __forceinline__ void async_load_b128(unsigned lds_addr, const void* gptr) {
  asm volatile("global_load_async_to_lds_b128 %0, %1, off"
               :: "v"(lds_addr), "v"(gptr)
               : "memory");
}
__device__ __forceinline__ void wait_asynccnt0() {
  asm volatile("s_wait_asynccnt 0x0" ::: "memory");
}

// ---------------------------------------------------------------------------
// GEMM: C[M=8192, N=512] = A[8192,512] x W[512,512] + bias
// Block tile 128x128, BK=32, 8 waves (4x2), wave tile 32x64 (2x4 WMMA tiles).
// LDS double-buffered: tile k+1 is staged (async for bf16 A, register-staged
// fp32->bf16 otherwise) while tile k's WMMAs execute; one barrier per step.
// SPLIT_HEAD: write bf16 output in [B,H,S,depth] layout; else fp32 row-major.
// ---------------------------------------------------------------------------
template<bool A_BF16, bool SPLIT_HEAD>
__global__ __launch_bounds__(256)
void gemm_wmma_512(const void* __restrict__ Ap,
                   const float* __restrict__ W,
                   const float* __restrict__ bias,
                   unsigned short* __restrict__ outB,
                   float* __restrict__ outF) {
  __shared__ unsigned short ldsA[2][128 * 40];   // 128x32 tiles, padded stride
  __shared__ unsigned short ldsB[2][32 * 136];   // 32x128 tiles, padded stride

  const int tid  = threadIdx.x;
  const int lane = tid & 31;
  const int wave = tid >> 5;
  const int wm   = wave >> 1;                 // 0..3
  const int wn   = wave & 1;                  // 0..1
  const int bm   = blockIdx.y * 128;
  const int bn   = blockIdx.x * 128;

  const float*          Af = (const float*)Ap;
  const unsigned short* Ab = (const unsigned short*)Ap;

  float4 aReg[4];                       // staged fp32 A tile (non-bf16 path)
  float4 bReg[4];                       // staged fp32 W tile

  auto issueA_async = [&](int b, int kk0) {
    #pragma unroll
    for (int i = 0; i < 2; ++i) {
      int g = tid + i * 256;            // 512 chunks of 8 elements
      int row = g >> 2, c8 = (g & 3) << 3;
      async_load_b128((unsigned)(size_t)&ldsA[b][row * 40 + c8],
                      &Ab[(size_t)(bm + row) * D_MODEL + kk0 + c8]);
    }
  };
  auto loadA_regs = [&](int kk0) {
    #pragma unroll
    for (int i = 0; i < 4; ++i) {
      int g = tid + i * 256;            // 1024 chunks of 4 elements
      int row = g >> 3, c4 = (g & 7) << 2;
      aReg[i] = *reinterpret_cast<const float4*>(
          &Af[(size_t)(bm + row) * D_MODEL + kk0 + c4]);
    }
  };
  auto storeA_lds = [&](int b) {
    #pragma unroll
    for (int i = 0; i < 4; ++i) {
      int g = tid + i * 256;
      int row = g >> 3, c4 = (g & 7) << 2;
      uint2 v; v.x = pack_bf16x2(aReg[i].x, aReg[i].y);
               v.y = pack_bf16x2(aReg[i].z, aReg[i].w);
      *reinterpret_cast<uint2*>(&ldsA[b][row * 40 + c4]) = v;
    }
  };
  auto loadB_regs = [&](int kk0) {
    #pragma unroll
    for (int i = 0; i < 4; ++i) {
      int g = tid + i * 256;
      int row = g >> 5, c4 = (g & 31) << 2;
      bReg[i] = *reinterpret_cast<const float4*>(
          &W[(size_t)(kk0 + row) * D_MODEL + bn + c4]);
    }
  };
  auto storeB_lds = [&](int b) {
    #pragma unroll
    for (int i = 0; i < 4; ++i) {
      int g = tid + i * 256;
      int row = g >> 5, c4 = (g & 31) << 2;
      uint2 v; v.x = pack_bf16x2(bReg[i].x, bReg[i].y);
               v.y = pack_bf16x2(bReg[i].z, bReg[i].w);
      *reinterpret_cast<uint2*>(&ldsB[b][row * 136 + c4]) = v;
    }
  };

  v8f acc[2][4] = {};

  // ---- prologue: stage tile 0 ----
  if constexpr (A_BF16) issueA_async(0, 0);
  else { loadA_regs(0); storeA_lds(0); }
  loadB_regs(0); storeB_lds(0);
  if constexpr (A_BF16) wait_asynccnt0();
  __syncthreads();

  #pragma unroll 2
  for (int kt = 0; kt < D_MODEL / 32; ++kt) {
    const int c  = kt & 1;
    const int kn = (kt + 1) * 32;

    if (kt < D_MODEL / 32 - 1) {        // begin staging tile k+1
      if constexpr (A_BF16) issueA_async(c ^ 1, kn);
      else loadA_regs(kn);
      loadB_regs(kn);
    }

    FragBF af[2];
    #pragma unroll
    for (int mf = 0; mf < 2; ++mf)
      af[mf] = load_frag_a(&ldsA[c][(wm * 32 + mf * 16) * 40], 40, lane);
    #pragma unroll
    for (int nf = 0; nf < 4; ++nf) {
      FragBF bfr = load_frag_b(&ldsB[c][wn * 64 + nf * 16], 136, lane);
      #pragma unroll
      for (int mf = 0; mf < 2; ++mf)
        acc[mf][nf] = wmma_bf16(af[mf], bfr, acc[mf][nf]);
    }

    if (kt < D_MODEL / 32 - 1) {        // land tile k+1 in the shadow buffer
      if constexpr (!A_BF16) storeA_lds(c ^ 1);
      storeB_lds(c ^ 1);
      if constexpr (A_BF16) wait_asynccnt0();
    }
    __syncthreads();
  }

  // ---- epilogue: C/D layout = VGPR i -> rows i (lanes 0-15) / 8+i (16-31) ----
  const int hlf  = lane >> 4;
  const int col0 = lane & 15;
  #pragma unroll
  for (int mf = 0; mf < 2; ++mf) {
    #pragma unroll
    for (int nf = 0; nf < 4; ++nf) {
      int   c  = bn + wn * 64 + nf * 16 + col0;
      float bv = bias[c];
      #pragma unroll
      for (int i = 0; i < 8; ++i) {
        int   r   = bm + wm * 32 + mf * 16 + i + 8 * hlf;
        float val = acc[mf][nf][i] + bv;
        if constexpr (SPLIT_HEAD) {
          int b = r >> 12, s = r & (SEQ - 1);
          int h = c >> 6,  d = c & 63;
          outB[((size_t)(b * NHEAD + h) * SEQ + s) * DEPTH + d] = f32_to_bf16(val);
        } else {
          outF[(size_t)r * D_MODEL + c] = val;
        }
      }
    }
  }
}

// ---------------------------------------------------------------------------
// Flash attention, operand-swapped: per tile computes S^T = K * Q^T so that
//  - K streams row-major via async global->LDS (no transpose anywhere),
//  - Q^T is built once per block and its B-fragments stay in registers,
//  - softmax rows (queries) collapse to per-lane scalars: reduce over the 8
//    accumulator VGPRs locally + one shfl_xor(16) across lane halves.
// K/V tiles are double-buffered in LDS; async fetch of tile j+1 overlaps the
// WMMAs of tile j. One barrier per iteration; loop unrolled x2 so the two
// buffer phases specialize.
// ---------------------------------------------------------------------------
__global__ __launch_bounds__(256, 2)
void flash_attn_kernel(const unsigned short* __restrict__ qp,
                       const unsigned short* __restrict__ kp,
                       const unsigned short* __restrict__ vp,
                       const float* __restrict__ maskp,
                       unsigned short* __restrict__ outp) {
  __shared__ unsigned short ldsK[2][64 * 72];    // [key][depth], padded
  __shared__ unsigned short ldsV[2][64 * 72];    // [key][depth], padded
  __shared__ unsigned short ldsP[8][16 * 72];    // per-wave P (query x key)
  __shared__ unsigned short ldsQT[8][64 * 24];   // per-wave Q^T [depth][query]

  const int tid  = threadIdx.x;
  const int lane = tid & 31;
  const int wave = tid >> 5;
  const int bh   = blockIdx.y;          // 0..15
  const int bidx = bh >> 3;
  const int h    = bh & 7;
  const int q0   = blockIdx.x * 128;
  const size_t headbase = (size_t)bh * SEQ * DEPTH;
  const float* mrow = maskp + (size_t)bidx * SEQ;
  const int hlf  = lane >> 4;
  const int col0 = lane & 15;

  // ---- issue async fetch of first K/V tile immediately ----
  #pragma unroll
  for (int i = 0; i < 2; ++i) {
    int g   = tid + i * 256;            // 512 chunks of 8 elements
    int key = g >> 3;
    int d8  = (g & 7) << 3;
    size_t goff = headbase + (size_t)key * DEPTH + d8;
    async_load_b128((unsigned)(size_t)&ldsK[0][key * 72 + d8], kp + goff);
    async_load_b128((unsigned)(size_t)&ldsV[0][key * 72 + d8], vp + goff);
  }

  // ---- stage Q^T per wave (loop-invariant) ----
  unsigned short* ldsQTw = ldsQT[wave];
  #pragma unroll
  for (int i = 0; i < 4; ++i) {
    int g   = lane + i * 32;            // 128 chunks of 8 elements
    int row = g >> 3;                   // query row 0..15
    int d8  = (g & 7) << 3;
    uint4 qq = *reinterpret_cast<const uint4*>(
        &qp[headbase + (size_t)(q0 + wave * 16 + row) * DEPTH + d8]);
    const unsigned short* qs = reinterpret_cast<const unsigned short*>(&qq);
    #pragma unroll
    for (int e = 0; e < 8; ++e) ldsQTw[(d8 + e) * 24 + row] = qs[e];
  }
  FragBF bqt[2];                        // Q^T B-fragments, live all iterations
  #pragma unroll
  for (int kc = 0; kc < 2; ++kc)
    bqt[kc] = load_frag_b(&ldsQTw[(kc * 32) * 24], 24, lane);

  v8f o[4] = {};
  float m_run = -3.0e38f, l_run = 0.0f;

  #pragma unroll 2
  for (int jb = 0; jb < SEQ / 64; ++jb) {
    const int buf = jb & 1;
    const int j0  = jb * 64;

    wait_asynccnt0();                   // this wave's tile-jb writes landed
    __syncthreads();                    // everyone's landed; buf^1 free to fill

    if (jb + 1 < SEQ / 64) {
      #pragma unroll
      for (int i = 0; i < 2; ++i) {
        int g   = tid + i * 256;
        int key = g >> 3;
        int d8  = (g & 7) << 3;
        size_t goff = headbase + (size_t)(j0 + 64 + key) * DEPTH + d8;
        async_load_b128((unsigned)(size_t)&ldsK[buf ^ 1][key * 72 + d8], kp + goff);
        async_load_b128((unsigned)(size_t)&ldsV[buf ^ 1][key * 72 + d8], vp + goff);
      }
    }

    const unsigned short* K = ldsK[buf];
    const unsigned short* V = ldsV[buf];

    // ---- S^T = K Q^T : element (key kf*16+i+8*hlf, query col0) per lane ----
    v8f sc[4];
    #pragma unroll
    for (int kf = 0; kf < 4; ++kf) {
      v8f s = {};
      #pragma unroll
      for (int kc = 0; kc < 2; ++kc) {
        FragBF ak = load_frag_a(&K[(kf * 16) * 72 + kc * 32], 72, lane);
        s = wmma_bf16(ak, bqt[kc], s);
      }
      // mask indexed by key (varies along accumulator VGPRs)
      const float4* mp = reinterpret_cast<const float4*>(&mrow[j0 + kf * 16 + 8 * hlf]);
      float4 m0 = mp[0], m1 = mp[1];
      s[0] += -1.0e9f * m0.x; s[1] += -1.0e9f * m0.y;
      s[2] += -1.0e9f * m0.z; s[3] += -1.0e9f * m0.w;
      s[4] += -1.0e9f * m1.x; s[5] += -1.0e9f * m1.y;
      s[6] += -1.0e9f * m1.z; s[7] += -1.0e9f * m1.w;
      sc[kf] = s;
    }

    // ---- per-query (= per-lane) online softmax ----
    float t = sc[0][0];
    #pragma unroll
    for (int kf = 0; kf < 4; ++kf)
      #pragma unroll
      for (int i = 0; i < 8; ++i) t = fmaxf(t, sc[kf][i]);
    t = fmaxf(t, __shfl_xor(t, 16, 32));
    float mn  = fmaxf(m_run, t);
    float sca = __expf(m_run - mn);
    m_run = mn;

    float rs = 0.0f;
    unsigned short* Pw = ldsP[wave];
    #pragma unroll
    for (int kf = 0; kf < 4; ++kf) {
      #pragma unroll
      for (int i = 0; i < 8; ++i) {
        float p = __expf(sc[kf][i] - mn);
        rs += p;
        // store transposed: row = query(col0), col = key
        Pw[col0 * 72 + kf * 16 + 8 * hlf + i] = f32_to_bf16(p);
      }
    }
    rs += __shfl_xor(rs, 16, 32);
    l_run = l_run * sca + rs;

    // ---- rescale O (row query = i + 8*hlf; scale lives in lane = query) ----
    #pragma unroll
    for (int i = 0; i < 8; ++i) {
      float si = __shfl(sca, i + 8 * hlf, 32);
      #pragma unroll
      for (int of = 0; of < 4; ++of) o[of][i] *= si;
    }

    // ---- O += P V ----
    FragBF ap[2];
    #pragma unroll
    for (int kc = 0; kc < 2; ++kc)
      ap[kc] = load_frag_a(&Pw[kc * 32], 72, lane);
    #pragma unroll
    for (int of = 0; of < 4; ++of) {
      #pragma unroll
      for (int kc = 0; kc < 2; ++kc) {
        FragBF bvf = load_frag_b(&V[(kc * 32) * 72 + of * 16], 72, lane);
        o[of] = wmma_bf16(ap[kc], bvf, o[of]);
      }
    }
    // no trailing barrier: next iteration's wait+barrier fences buffer reuse
  }

  // ---- epilogue: O / l, write bf16 [B,S,H*depth] for the output GEMM ----
  #pragma unroll
  for (int i = 0; i < 8; ++i) {
    float li  = __shfl(l_run, i + 8 * hlf, 32);
    float inv = 1.0f / li;
    int s = q0 + wave * 16 + i + 8 * hlf;
    size_t ob = ((size_t)bidx * SEQ + s) * D_MODEL + h * DEPTH;
    #pragma unroll
    for (int of = 0; of < 4; ++of)
      outp[ob + of * 16 + col0] = f32_to_bf16(o[of][i] * inv);
  }
}

// ---------------------------------------------------------------------------
extern "C" void kernel_launch(void* const* d_in, const int* in_sizes, int n_in,
                              void* d_out, int out_size, void* d_ws, size_t ws_size,
                              hipStream_t stream) {
  (void)in_sizes; (void)n_in; (void)out_size; (void)ws_size;
  const float* q    = (const float*)d_in[0];
  const float* k    = (const float*)d_in[1];
  const float* v    = (const float*)d_in[2];
  const float* mask = (const float*)d_in[3];
  const float* wq   = (const float*)d_in[4];
  const float* bq   = (const float*)d_in[5];
  const float* wk   = (const float*)d_in[6];
  const float* bk   = (const float*)d_in[7];
  const float* wv   = (const float*)d_in[8];
  const float* bv   = (const float*)d_in[9];
  const float* wo   = (const float*)d_in[10];
  const float* bo   = (const float*)d_in[11];

  const size_t T = (size_t)BATCH * NHEAD * SEQ * DEPTH;  // 4,194,304 elems
  unsigned short* qp = (unsigned short*)d_ws;
  unsigned short* kp = qp + T;
  unsigned short* vp = kp + T;
  unsigned short* ao = vp + T;   // attention output, bf16 [B,S,D]

  dim3 ggrid(D_MODEL / 128, (BATCH * SEQ) / 128);   // (4, 64)

  gemm_wmma_512<false, true><<<ggrid, 256, 0, stream>>>(q, wq, bq, qp, nullptr);
  gemm_wmma_512<false, true><<<ggrid, 256, 0, stream>>>(k, wk, bk, kp, nullptr);
  gemm_wmma_512<false, true><<<ggrid, 256, 0, stream>>>(v, wv, bv, vp, nullptr);

  flash_attn_kernel<<<dim3(SEQ / 128, BATCH * NHEAD), 256, 0, stream>>>(
      qp, kp, vp, mask, ao);

  gemm_wmma_512<true, false><<<ggrid, 256, 0, stream>>>(ao, wo, bo, nullptr,
                                                        (float*)d_out);
}
